// TransformerBlock_2937757631133
// MI455X (gfx1250) — matrix operand order repaired and checked by reading, not compiled, not run on hardware
//
#include <hip/hip_runtime.h>
#include <math.h>

typedef _Float16 h16;
typedef __attribute__((ext_vector_type(16))) _Float16 v16h;
typedef __attribute__((ext_vector_type(8)))  _Float16 v8h;
typedef __attribute__((ext_vector_type(8)))  float    v8f;
typedef __attribute__((ext_vector_type(4)))  int      v4i;

#define AS1 __attribute__((address_space(1)))
#define AS3 __attribute__((address_space(3)))

#if __has_builtin(__builtin_amdgcn_global_load_async_to_lds_b128)
#define USE_ASYNC_LDS 1
#else
#define USE_ASYNC_LDS 0
#endif

// ---------------------------------------------------------------------------
// CDNA5 helpers (gfx1250, wave32)
// ---------------------------------------------------------------------------
__device__ __forceinline__ v8f wmma_f16(v16h a, v16h b, v8f c) {
  // D(16x16 f32) = A(16x32 f16) * B(32x16 f16) + C
  return __builtin_amdgcn_wmma_f32_16x16x32_f16(false, a, false, b, (short)0, c,
                                                false, false);
}

// Force-global pointer (keeps staging loads as global_load_*, not flat_load_*,
// so DScnt stays exclusive to LDS traffic per CDNA5 counter rules).
__device__ __forceinline__ const AS1 h16* as_g(const h16* p) {
  return (const AS1 h16*)(uintptr_t)p;
}

// 16B global -> LDS, tracked by ASYNCcnt (falls back to sync copy if the
// builtin is unavailable on this toolchain).
__device__ __forceinline__ void async_g2l_b128(const AS1 h16* gsrc, h16* ldst) {
#if USE_ASYNC_LDS
  __builtin_amdgcn_global_load_async_to_lds_b128(
      (AS1 v4i*)(uintptr_t)gsrc,
      (AS3 v4i*)(unsigned)(uintptr_t)ldst, 0, 0);
#else
  *(v8h*)ldst = *(const AS1 v8h*)gsrc;
#endif
}

__device__ __forceinline__ void async_wait0() {
#if USE_ASYNC_LDS
#if __has_builtin(__builtin_amdgcn_s_wait_asynccnt)
  __builtin_amdgcn_s_wait_asynccnt(0);
#else
  asm volatile("s_wait_asynccnt 0" ::: "memory");
#endif
#endif
}

// A fragment: lane holds row M = lane&15; k elements = base + {0..7} and
// base + {16..23}, base = (lane>=16 ? 8 : 0) folded into p0 by the caller.
__device__ __forceinline__ v16h frag_a_from(const h16* p0) {
  v16h a;
#pragma unroll
  for (int e = 0; e < 8; ++e) a[e] = p0[e];
#pragma unroll
  for (int e = 8; e < 16; ++e) a[e] = p0[e + 8];
  return a;
}

// B fragment: lane holds col N = lane&15; k elements = base + {0..7} and
// base + {16..23}, base = (lane>=16 ? 8 : 0) folded into p0 (requires K-contiguous storage).
__device__ __forceinline__ v16h frag_b_from(const h16* p0) {
  v16h b;
#pragma unroll
  for (int e = 0; e < 8; ++e) b[e] = p0[e];
#pragma unroll
  for (int e = 8; e < 16; ++e) b[e] = p0[e + 8];
  return b;
}

// ---------------------------------------------------------------------------
// Generic f16 WMMA GEMM:  C(MxN) = A(MxK) * B(KxN), fused epilogues.
// Block = 128 threads (4 waves), tile 128x128, K staged 32 at a time in LDS.
// MODE 0: scatter to q/k/v [B,H,T,64] f16 (q scaled by 0.125)
// MODE 1: out_f32 = acc + bias[col] + res      (f32 row-major MxN)
// MODE 2: out_f16 = gelu(acc + bias[col])      (f16 row-major MxN)
// Templated on MODE so each epilogue is small and branch-free (avoids the
// accumulator scratch spill seen with a runtime mode switch).
// ---------------------------------------------------------------------------
#define GPITCH 72   // halves per LDS row (32 data + pad), rows 16B aligned

template <int MODE>
__launch_bounds__(128, 1)
__global__ void gemm_wmma(const h16* __restrict__ A, const h16* __restrict__ B,
                          int N, int K,
                          const float* __restrict__ bias,
                          const float* __restrict__ res,
                          float* __restrict__ outf, h16* __restrict__ outh,
                          h16* __restrict__ qd, h16* __restrict__ kd,
                          h16* __restrict__ vd)
{
  __shared__ h16 sA[128 * GPITCH];   // [m][k]
  __shared__ h16 sB[128 * GPITCH];   // [n][k]  (transposed: k contiguous)
  const int tid  = threadIdx.x;
  const int lane = tid & 31, wid = tid >> 5;
  const int ln   = lane & 15, lh = lane >> 4;
  const int wm   = (wid >> 1) * 64, wn = (wid & 1) * 64;
  const int m0   = blockIdx.y * 128, n0 = blockIdx.x * 128;

  // hoisted per-thread staging addresses (bumped by constant strides),
  // kept in addrspace(1) so they lower to global_load_* even through PHIs
  const AS1 h16* aSrc[4]; const AS1 h16* bSrc[4];
  h16* aDst[4]; int bKr[4], bN8[4];
#pragma unroll
  for (int i = 0; i < 4; ++i) {
    int c = tid + i * 128;
    int ar = c >> 2, ac8 = (c & 3) << 3;
    aSrc[i] = as_g(A + (size_t)(m0 + ar) * K + ac8);
    aDst[i] = &sA[ar * GPITCH + ac8];
    bKr[i] = c >> 4; bN8[i] = (c & 15) << 3;
    bSrc[i] = as_g(B + (size_t)bKr[i] * N + n0 + bN8[i]);
  }

  const v8f zf = {0.f,0.f,0.f,0.f,0.f,0.f,0.f,0.f};
  v8f acc[4][4];
#pragma unroll
  for (int i = 0; i < 4; ++i)
#pragma unroll
    for (int j = 0; j < 4; ++j) acc[i][j] = zf;

  for (int kb = 0; kb < K; kb += 32) {
    // stage A tile 128x32 via async memory->LDS (ASYNCcnt)
#pragma unroll
    for (int i = 0; i < 4; ++i) {
      async_g2l_b128(aSrc[i], aDst[i]);
      aSrc[i] += 32;
    }
    // stage B tile 32x128 transposed into [n][k] (register transpose)
#pragma unroll
    for (int i = 0; i < 4; ++i) {
      v8h t = *(const AS1 v8h*)bSrc[i];
      bSrc[i] += (size_t)32 * N;
#pragma unroll
      for (int j = 0; j < 8; ++j) sB[(bN8[i] + j) * GPITCH + bKr[i]] = t[j];
    }
    async_wait0();
    __syncthreads();

    v16h af[4];
#pragma unroll
    for (int i = 0; i < 4; ++i)
      af[i] = frag_a_from(&sA[(wm + i * 16 + ln) * GPITCH + (lh << 3)]);
#pragma unroll
    for (int j = 0; j < 4; ++j) {
      v16h bf = frag_b_from(&sB[(wn + j * 16 + ln) * GPITCH + (lh << 3)]);
#pragma unroll
      for (int i = 0; i < 4; ++i) acc[i][j] = wmma_f16(af[i], bf, acc[i][j]);
    }
    __syncthreads();
  }

  // epilogue (C layout: lane = col, elem e = row lh*8+e)
#pragma unroll
  for (int i = 0; i < 4; ++i)
#pragma unroll
    for (int j = 0; j < 4; ++j) {
      const int gc = n0 + wn + j * 16 + ln;
#pragma unroll
      for (int e = 0; e < 8; ++e) {
        const int gm = m0 + wm + i * 16 + lh * 8 + e;
        const float av = acc[i][j][e];
        if (MODE == 1) {
          outf[(size_t)gm * N + gc] = av + bias[gc] + res[(size_t)gm * N + gc];
        } else if (MODE == 2) {
          float z = av + bias[gc];
          outh[(size_t)gm * N + gc] =
              (h16)(0.5f * z * (1.0f + erff(z * 0.70710678118654752f)));
        } else {  // QKV scatter: col -> (which, head, kdim); row -> (b, t)
          int which = gc >> 10, w = gc & 1023, hh = w >> 6, kdim = w & 63;
          int b_ = gm >> 11, t_ = gm & 2047;
          size_t di = (((size_t)(b_ * 16 + hh)) * 2048 + t_) * 64 + kdim;
          if (which == 0)       qd[di] = (h16)(av * 0.125f);  // fold 1/sqrt(64)
          else if (which == 1)  kd[di] = (h16)av;
          else                  vd[di] = (h16)av;
        }
      }
    }
}

// ---------------------------------------------------------------------------
// Flash attention (causal): grid (T/64, B*H), 128 threads = 4 waves,
// wave handles 16 query rows; K and V^T tiles (64x64 f16) staged in LDS.
// q is pre-scaled by 1/sqrt(64). Output written as [b, t, h*64+kk] f16.
// ---------------------------------------------------------------------------
__launch_bounds__(128, 1)
__global__ void attn_wmma(const h16* __restrict__ q, const h16* __restrict__ kmat,
                          const h16* __restrict__ vmat, h16* __restrict__ o)
{
  constexpr int P = 72;
  __shared__ h16 sK [64 * P];      // [s][kdim]  -> B-frag for Q*K^T
  __shared__ h16 sVt[64 * P];      // [kdim][s]  -> B-frag for P*V
  __shared__ h16 sP [4 * 16 * P];  // per-wave P tile [m][s]
  const int tid = threadIdx.x, lane = tid & 31, wid = tid >> 5;
  const int ln = lane & 15, lh = lane >> 4;
  const int bh = blockIdx.y;                 // b*16 + h
  const int tq0 = blockIdx.x * 64;
  const int tqw = tq0 + wid * 16;            // this wave's first query row
  const size_t base = (size_t)bh * 2048 * 64;

  // Q fragments straight from global (A layout, 2 k-steps of 32)
  const AS1 h16* qp = as_g(q + base + (size_t)(tqw + ln) * 64 + (lh << 3));
  v16h qf[2];
#pragma unroll
  for (int ks = 0; ks < 2; ++ks) {
#pragma unroll
    for (int e = 0; e < 8; ++e)  qf[ks][e] = qp[ks * 32 + e];
#pragma unroll
    for (int e = 8; e < 16; ++e) qf[ks][e] = qp[ks * 32 + e + 8];
  }

  // hoisted per-thread staging addresses for the chunk loop (addrspace(1))
  const AS1 h16* kSrc[4]; const AS1 h16* vSrc[4];
  h16* kDst[4]; int vSr[4], vC8[4];
#pragma unroll
  for (int i = 0; i < 4; ++i) {
    int c = tid + i * 128;
    int sr = c >> 3, c8 = (c & 7) << 3;
    kSrc[i] = as_g(kmat + base + (size_t)sr * 64 + c8);
    vSrc[i] = as_g(vmat + base + (size_t)sr * 64 + c8);
    kDst[i] = &sK[sr * P + c8];
    vSr[i] = sr; vC8[i] = c8;
  }

  const v8f zf = {0.f,0.f,0.f,0.f,0.f,0.f,0.f,0.f};
  v8f oacc[4];
  float rmax[8], rsum[8];
#pragma unroll
  for (int j = 0; j < 4; ++j) oacc[j] = zf;
#pragma unroll
  for (int e = 0; e < 8; ++e) { rmax[e] = -3.0e30f; rsum[e] = 0.f; }

  h16* sPw = &sP[wid * 16 * P];
  const int nchunk = blockIdx.x + 1;
  for (int ch = 0; ch < nchunk; ++ch) {
    const int s0 = ch * 64;
    // K tile async to LDS; V tile register-transposed into [kdim][s]
#pragma unroll
    for (int i = 0; i < 4; ++i) {
      async_g2l_b128(kSrc[i], kDst[i]);
      kSrc[i] += 4096;
      v8h tv = *(const AS1 v8h*)vSrc[i];
      vSrc[i] += 4096;
#pragma unroll
      for (int j = 0; j < 8; ++j) sVt[(vC8[i] + j) * P + vSr[i]] = tv[j];
    }
    async_wait0();
    __syncthreads();

    if (s0 <= tqw + 15) {            // uniform per wave
      // scores S = Q * K^T  (16 x 64)
      v8f sacc[4];
#pragma unroll
      for (int j = 0; j < 4; ++j) sacc[j] = zf;
#pragma unroll
      for (int ks = 0; ks < 2; ++ks)
#pragma unroll
        for (int j = 0; j < 4; ++j)
          sacc[j] = wmma_f16(qf[ks],
              frag_b_from(&sK[(j * 16 + ln) * P + ks * 32 + (lh << 3)]),
              sacc[j]);

      const bool full = (s0 + 63 <= tqw);
      float pv[4][8];
#pragma unroll
      for (int e = 0; e < 8; ++e) {
        const int trow = tqw + lh * 8 + e;
        float mloc = -3.0e30f;
#pragma unroll
        for (int j = 0; j < 4; ++j) {
          float sv = sacc[j][e];
          if (!full && (s0 + j * 16 + ln) > trow) sv = -3.0e30f;
          pv[j][e] = sv;
          mloc = fmaxf(mloc, sv);
        }
#pragma unroll
        for (int off = 8; off >= 1; off >>= 1)
          mloc = fmaxf(mloc, __shfl_xor(mloc, off, 32));
        const float mnew  = fmaxf(rmax[e], mloc);
        const float scale = __expf(rmax[e] - mnew);
        float ls = 0.f;
#pragma unroll
        for (int j = 0; j < 4; ++j) {
          float p = __expf(pv[j][e] - mnew);
          pv[j][e] = p;
          ls += p;
        }
#pragma unroll
        for (int off = 8; off >= 1; off >>= 1) ls += __shfl_xor(ls, off, 32);
        rsum[e] = rsum[e] * scale + ls;
        rmax[e] = mnew;
#pragma unroll
        for (int j = 0; j < 4; ++j) oacc[j][e] = oacc[j][e] * scale;
        // spill P to LDS (C layout -> [m][s]) for A-frag reload
#pragma unroll
        for (int j = 0; j < 4; ++j)
          sPw[(lh * 8 + e) * P + j * 16 + ln] = (h16)pv[j][e];
      }
      asm volatile("s_wait_dscnt 0" ::: "memory");  // same-wave LDS RAW

      // O += P(16x64) * V(64x64)
#pragma unroll
      for (int ks = 0; ks < 2; ++ks) {
        v16h af = frag_a_from(&sPw[ln * P + ks * 32 + (lh << 3)]);
#pragma unroll
        for (int j = 0; j < 4; ++j)
          oacc[j] = wmma_f16(af,
              frag_b_from(&sVt[(j * 16 + ln) * P + ks * 32 + (lh << 3)]),
              oacc[j]);
      }
    }
    __syncthreads();
  }

  // normalize and store heads-concat layout [b, t, h*64 + kk]
  const int b_ = bh >> 4, h_ = bh & 15;
#pragma unroll
  for (int j = 0; j < 4; ++j)
#pragma unroll
    for (int e = 0; e < 8; ++e) {
      const int trow = tqw + lh * 8 + e;
      o[((size_t)(b_ * 2048 + trow)) * 1024 + h_ * 64 + j * 16 + ln] =
          (h16)(oacc[j][e] / rsum[e]);
    }
}

// ---------------------------------------------------------------------------
// LayerNorm (one 256-thread block per row of 1024), f32 in, f16 out
// ---------------------------------------------------------------------------
__launch_bounds__(256)
__global__ void ln_f16(const float* __restrict__ x, const float* __restrict__ g,
                       const float* __restrict__ be, h16* __restrict__ out)
{
  __shared__ float red0[8], red1[8];
  const int row = blockIdx.x, tid = threadIdx.x;
  const float* xr = x + (size_t)row * 1024;
  float v4[4], s = 0.f, s2 = 0.f;
#pragma unroll
  for (int i = 0; i < 4; ++i) {
    float xv = xr[tid + i * 256];
    v4[i] = xv; s += xv; s2 += xv * xv;
  }
#pragma unroll
  for (int off = 16; off >= 1; off >>= 1) {
    s += __shfl_xor(s, off, 32);  s2 += __shfl_xor(s2, off, 32);
  }
  if ((tid & 31) == 0) { red0[tid >> 5] = s; red1[tid >> 5] = s2; }
  __syncthreads();
  float ts = 0.f, ts2 = 0.f;
#pragma unroll
  for (int i = 0; i < 8; ++i) { ts += red0[i]; ts2 += red1[i]; }
  const float mu = ts * (1.0f / 1024.0f);
  const float var = ts2 * (1.0f / 1024.0f) - mu * mu;
  const float rs = rsqrtf(var + 1e-5f);
#pragma unroll
  for (int i = 0; i < 4; ++i) {
    int c = tid + i * 256;
    out[(size_t)row * 1024 + c] = (h16)((v4[i] - mu) * rs * g[c] + be[c]);
  }
}

// ---------------------------------------------------------------------------
// Weight conversion kernels
// ---------------------------------------------------------------------------
__global__ void cvt_f16(const float* __restrict__ s, h16* __restrict__ d, int n) {
  int i = blockIdx.x * 256 + threadIdx.x;
  if (i < n) d[i] = (h16)s[i];
}

// Wq/Wk/Wv are [H=16, D=1024, K=64]; pack to Wcat f16 [D, 3*1024] with column
// order (which, head, kdim) so the QKV GEMM epilogue can scatter per head.
__global__ void pack_qkv(const float* __restrict__ Wq, const float* __restrict__ Wk,
                         const float* __restrict__ Wv, h16* __restrict__ Wcat) {
  int i = blockIdx.x * 256 + threadIdx.x;
  if (i >= 1024 * 3072) return;
  int c = i % 3072, d = i / 3072;
  int which = c >> 10, w = c & 1023, hh = w >> 6, kdim = w & 63;
  const float* W = (which == 0) ? Wq : (which == 1) ? Wk : Wv;
  Wcat[i] = (h16)W[((size_t)hh * 1024 + d) * 64 + kdim];
}

// ---------------------------------------------------------------------------
// Launch: LN1 -> QKV GEMM -> flash attn -> out-proj(+res) -> LN2 -> FF1(GELU)
//         -> FF2(+res) ; all WMMA GEMMs, all on `stream`.
// ---------------------------------------------------------------------------
extern "C" void kernel_launch(void* const* d_in, const int* in_sizes, int n_in,
                              void* d_out, int out_size, void* d_ws, size_t ws_size,
                              hipStream_t stream)
{
  (void)in_sizes; (void)n_in; (void)out_size; (void)ws_size;
  const float* x   = (const float*)d_in[0];
  const float* Wq  = (const float*)d_in[1];
  const float* Wk  = (const float*)d_in[2];
  const float* Wv  = (const float*)d_in[3];
  const float* Wo  = (const float*)d_in[4];
  const float* bo  = (const float*)d_in[5];
  const float* W1  = (const float*)d_in[6];
  const float* b1  = (const float*)d_in[7];
  const float* W2  = (const float*)d_in[8];
  const float* b2  = (const float*)d_in[9];
  const float* g1  = (const float*)d_in[10];
  const float* be1 = (const float*)d_in[11];
  const float* g2  = (const float*)d_in[12];
  const float* be2 = (const float*)d_in[13];
  float* out = (float*)d_out;

  const int BT = 4096;  // B*T
  char* ws = (char*)d_ws;
  size_t off = 0;
  auto carve = [&](size_t bytes) -> void* {
    void* p = ws + off; off += (bytes + 255) & ~(size_t)255; return p;
  };
  h16* h1   = (h16*)carve((size_t)BT * 1024 * 2);
  h16* h2   = (h16*)carve((size_t)BT * 1024 * 2);
  h16* qb   = (h16*)carve((size_t)BT * 1024 * 2);
  h16* kbuf = (h16*)carve((size_t)BT * 1024 * 2);
  h16* vb   = (h16*)carve((size_t)BT * 1024 * 2);
  h16* ob   = (h16*)carve((size_t)BT * 1024 * 2);
  float* x1 = (float*)carve((size_t)BT * 1024 * 4);
  h16* a1   = (h16*)carve((size_t)BT * 4096 * 2);
  h16* Wcat = (h16*)carve((size_t)1024 * 3072 * 2);
  h16* Wo16 = (h16*)carve((size_t)1024 * 1024 * 2);
  h16* W116 = (h16*)carve((size_t)1024 * 4096 * 2);
  h16* W216 = (h16*)carve((size_t)4096 * 1024 * 2);

  dim3 b256(256), b128(128);
  pack_qkv<<<dim3((1024 * 3072 + 255) / 256), b256, 0, stream>>>(Wq, Wk, Wv, Wcat);
  cvt_f16 <<<dim3((1024 * 1024 + 255) / 256), b256, 0, stream>>>(Wo, Wo16, 1024 * 1024);
  cvt_f16 <<<dim3((1024 * 4096 + 255) / 256), b256, 0, stream>>>(W1, W116, 1024 * 4096);
  cvt_f16 <<<dim3((4096 * 1024 + 255) / 256), b256, 0, stream>>>(W2, W216, 4096 * 1024);

  ln_f16<<<dim3(BT), b256, 0, stream>>>(x, g1, be1, h1);

  gemm_wmma<0><<<dim3(3072 / 128, BT / 128), b128, 0, stream>>>(
      h1, Wcat, 3072, 1024,
      nullptr, nullptr, nullptr, nullptr, qb, kbuf, vb);

  attn_wmma<<<dim3(2048 / 64, 32), b128, 0, stream>>>(qb, kbuf, vb, ob);

  gemm_wmma<1><<<dim3(1024 / 128, BT / 128), b128, 0, stream>>>(
      ob, Wo16, 1024, 1024,
      bo, x, x1, nullptr, nullptr, nullptr, nullptr);

  ln_f16<<<dim3(BT), b256, 0, stream>>>(x1, g2, be2, h2);

  gemm_wmma<2><<<dim3(4096 / 128, BT / 128), b128, 0, stream>>>(
      h2, W116, 4096, 1024,
      b1, nullptr, nullptr, a1, nullptr, nullptr, nullptr);

  gemm_wmma<1><<<dim3(1024 / 128, BT / 128), b128, 0, stream>>>(
      a1, W216, 1024, 4096,
      b2, x1, out, nullptr, nullptr, nullptr, nullptr);
}
